// GAT_24876450578592
// MI455X (gfx1250) — compile-verified, matrix operand order checked by reading, not run on hardware
//
#include <hip/hip_runtime.h>
#include <hip/hip_bf16.h>

#define SBATCH 2048
#define PASTN  64
#define FUTN   16

typedef __attribute__((ext_vector_type(16))) __bf16 v16bf;
typedef __attribute__((ext_vector_type(8)))  float  v8f;

union Frag16 { v16bf v; unsigned int u[8]; };

// ---- WMMA fragment helpers (bf16, 16x16x32, wave32) -------------------------
// A 16x32 bf16 per ISA 7.12.2: lane = row (lane&15); K packing:
//   element j: K = (j<8?0:16) + 8*(lane>>4) + (j&7)
__device__ inline v16bf lds_afrag(const __bf16* base, int strideE, int kstep, int lane) {
    const unsigned int* p = (const unsigned int*)base;
    int row = lane & 15;
    int kb  = ((lane >> 4) << 3) + (kstep << 5);
    int d   = (row * strideE + kb) >> 1;           // dword index (k even)
    Frag16 f;
#pragma unroll
    for (int t = 0; t < 4; ++t) { f.u[t] = p[d + t]; f.u[4 + t] = p[d + 8 + t]; }
    return f.v;
}

// B fragments pre-swizzled at staging time: blob[tile][lane][j] contiguous.
__device__ inline v16bf lds_bfrag(const __bf16* w, int tile, int lane) {
    const unsigned int* p = (const unsigned int*)w;
    int d = tile * 256 + lane * 8;
    Frag16 f;
#pragma unroll
    for (int t = 0; t < 8; ++t) f.u[t] = p[d + t];
    return f.v;
}

__device__ inline v8f wmma_bf16(v16bf a, v16bf b, v8f c) {
    return __builtin_amdgcn_wmma_f32_16x16x32_bf16(false, a, false, b, (short)0, c, false, false);
}

__device__ inline int kmap(int l, int j) {
    return ((j & 8) << 1) + ((l >> 4) << 3) + (j & 7);
}

// ---- Fused 3-layer pre-MLP:  gather-embed -> 35->128->128->34 (+num tail) ---
__global__ __launch_bounds__(256) void k_mlp(
    const int* __restrict__ cat, const float* __restrict__ num,
    const float* __restrict__ emb0, const float* __restrict__ emb1, const float* __restrict__ emb2,
    const float* __restrict__ W1, const float* __restrict__ b1,
    const float* __restrict__ W2, const float* __restrict__ b2,
    const float* __restrict__ W3, const float* __restrict__ b3,
    float* __restrict__ xpre, int Mrows)
{
    __shared__ __bf16 sW1[2 * 8 * 512];
    __shared__ __bf16 sW2[4 * 8 * 512];
    __shared__ __bf16 sW3[4 * 3 * 512];
    __shared__ float  sb1[128], sb2[128], sb3[48];
    __shared__ __bf16 act[8][16 * 128];

    int tid  = threadIdx.x;
    int lane = tid & 31;
    int wave = tid >> 5;

    // stage weights (fp32 -> bf16), pre-swizzled to B-fragment order
    for (int e = tid; e < 2 * 8 * 512; e += 256) {
        int tile = e >> 9, kt = tile >> 3, nt = tile & 7;
        int l = (e >> 4) & 31, j = e & 15;
        int k = kt * 32 + kmap(l, j);
        int n = nt * 16 + (l & 15);
        sW1[e] = (__bf16)((k < 35) ? W1[k * 128 + n] : 0.f);
    }
    for (int e = tid; e < 4 * 8 * 512; e += 256) {
        int tile = e >> 9, kt = tile >> 3, nt = tile & 7;
        int l = (e >> 4) & 31, j = e & 15;
        int k = kt * 32 + kmap(l, j);
        int n = nt * 16 + (l & 15);
        sW2[e] = (__bf16)W2[k * 128 + n];
    }
    for (int e = tid; e < 4 * 3 * 512; e += 256) {
        int tile = e >> 9, kt = tile / 3, nt = tile % 3;
        int l = (e >> 4) & 31, j = e & 15;
        int k = kt * 32 + kmap(l, j);
        int n = nt * 16 + (l & 15);
        sW3[e] = (__bf16)((n < 34) ? W3[k * 34 + n] : 0.f);
    }
    if (tid < 128) { sb1[tid] = b1[tid]; sb2[tid] = b2[tid]; }
    if (tid < 48)  sb3[tid] = (tid < 34) ? b3[tid] : 0.f;

    // build padded input features (16 rows x 64 k, stored stride 128) per wave
    int rowBase  = blockIdx.x * 128 + wave * 16;
    int rowLocal = lane & 15;
    int row      = rowBase + rowLocal;
    bool valid   = row < Mrows;
    int c0 = 0, c1 = 0, c2 = 0; float nm0 = 0, nm1 = 0, nm2 = 0;
    if (valid) {
        c0 = cat[row * 3 + 0]; c1 = cat[row * 3 + 1]; c2 = cat[row * 3 + 2];
        nm0 = num[row * 3 + 0]; nm1 = num[row * 3 + 1]; nm2 = num[row * 3 + 2];
    }
    int half = lane >> 4;
    for (int kk = 0; kk < 32; ++kk) {
        int k = half * 32 + kk;
        float v = 0.f;
        if (valid) {
            if (k < 16)      v = emb0[c0 * 16 + k];
            else if (k < 24) v = emb1[c1 * 8 + (k - 16)];
            else if (k < 32) v = emb2[c2 * 8 + (k - 24)];
            else if (k == 32) v = nm0;
            else if (k == 33) v = nm1;
            else if (k == 34) v = nm2;
        }
        act[wave][rowLocal * 128 + k] = (__bf16)v;
    }
    __syncthreads();

    v8f zero = {0.f, 0.f, 0.f, 0.f, 0.f, 0.f, 0.f, 0.f};
    int mofs = (lane >> 4) << 3;
    int n0   = lane & 15;

    // ---- layer 1: K=64(pad), N=128 ----
    {
        v8f acc[8];
#pragma unroll
        for (int nt = 0; nt < 8; ++nt) acc[nt] = zero;
#pragma unroll
        for (int ks = 0; ks < 2; ++ks) {
            v16bf a = lds_afrag(&act[wave][0], 128, ks, lane);
#pragma unroll
            for (int nt = 0; nt < 8; ++nt)
                acc[nt] = wmma_bf16(a, lds_bfrag(sW1, ks * 8 + nt, lane), acc[nt]);
        }
        __syncthreads();
#pragma unroll
        for (int nt = 0; nt < 8; ++nt)
#pragma unroll
            for (int j = 0; j < 8; ++j) {
                int col = nt * 16 + n0;
                float v = fmaxf(acc[nt][j] + sb1[col], 0.f);
                act[wave][(mofs + j) * 128 + col] = (__bf16)v;
            }
        __syncthreads();
    }
    // ---- layer 2: K=128, N=128 ----
    {
        v8f acc[8];
#pragma unroll
        for (int nt = 0; nt < 8; ++nt) acc[nt] = zero;
#pragma unroll
        for (int ks = 0; ks < 4; ++ks) {
            v16bf a = lds_afrag(&act[wave][0], 128, ks, lane);
#pragma unroll
            for (int nt = 0; nt < 8; ++nt)
                acc[nt] = wmma_bf16(a, lds_bfrag(sW2, ks * 8 + nt, lane), acc[nt]);
        }
        __syncthreads();
#pragma unroll
        for (int nt = 0; nt < 8; ++nt)
#pragma unroll
            for (int j = 0; j < 8; ++j) {
                int col = nt * 16 + n0;
                float v = fmaxf(acc[nt][j] + sb2[col], 0.f);
                act[wave][(mofs + j) * 128 + col] = (__bf16)v;
            }
        __syncthreads();
    }
    // ---- layer 3: K=128, N=48(pad, real 34) ----
    {
        v8f acc[3];
#pragma unroll
        for (int nt = 0; nt < 3; ++nt) acc[nt] = zero;
#pragma unroll
        for (int ks = 0; ks < 4; ++ks) {
            v16bf a = lds_afrag(&act[wave][0], 128, ks, lane);
#pragma unroll
            for (int nt = 0; nt < 3; ++nt)
                acc[nt] = wmma_bf16(a, lds_bfrag(sW3, ks * 3 + nt, lane), acc[nt]);
        }
#pragma unroll
        for (int nt = 0; nt < 3; ++nt)
#pragma unroll
            for (int j = 0; j < 8; ++j) {
                int col = nt * 16 + n0;
                if (col < 34) {
                    int r = rowBase + mofs + j;
                    if (r < Mrows) xpre[(size_t)r * 35 + col] = acc[nt][j] + sb3[col];
                }
            }
        if (lane < 16 && valid) xpre[(size_t)row * 35 + 34] = nm2;  // num last channel
    }
}

// ---- Generic WMMA GEMM:  C[M x N] = (RELU?relu:id)(A[M x K]) @ W[K x N] -----
template <int KT, int NT, int RELU>
__global__ __launch_bounds__(256) void k_gemm(
    const float* __restrict__ Ain, int M, int K, int ldA,
    const float* __restrict__ W, int N,
    float* __restrict__ Cout, int ldC)
{
    constexpr int STRIDE = KT * 32;
    __shared__ __bf16 sW[KT * NT * 512];
    __shared__ __bf16 act[8][16 * STRIDE];

    int tid = threadIdx.x, lane = tid & 31, wave = tid >> 5;

    for (int e = tid; e < KT * NT * 512; e += 256) {
        int tile = e >> 9, kt = tile / NT, nt = tile % NT;
        int l = (e >> 4) & 31, j = e & 15;
        int k = kt * 32 + kmap(l, j);
        int n = nt * 16 + (l & 15);
        sW[e] = (__bf16)((k < K && n < N) ? W[(size_t)k * N + n] : 0.f);
    }

    int rowBase  = blockIdx.x * 128 + wave * 16;
    int rowLocal = lane & 15;
    int row      = rowBase + rowLocal;
    int half     = lane >> 4;
    for (int kk = 0; kk < STRIDE / 2; ++kk) {
        int k = half * (STRIDE / 2) + kk;
        float v = 0.f;
        if (row < M && k < K) {
            v = Ain[(size_t)row * ldA + k];
            if (RELU) v = fmaxf(v, 0.f);
        }
        act[wave][rowLocal * STRIDE + k] = (__bf16)v;
    }
    __syncthreads();

    v8f zero = {0.f, 0.f, 0.f, 0.f, 0.f, 0.f, 0.f, 0.f};
    v8f acc[NT];
#pragma unroll
    for (int nt = 0; nt < NT; ++nt) acc[nt] = zero;
#pragma unroll
    for (int ks = 0; ks < KT; ++ks) {
        v16bf a = lds_afrag(&act[wave][0], STRIDE, ks, lane);
#pragma unroll
        for (int nt = 0; nt < NT; ++nt)
            acc[nt] = wmma_bf16(a, lds_bfrag(sW, ks * NT + nt, lane), acc[nt]);
    }
    int mofs = (lane >> 4) << 3, n0 = lane & 15;
#pragma unroll
    for (int nt = 0; nt < NT; ++nt)
#pragma unroll
        for (int j = 0; j < 8; ++j) {
            int col = nt * 16 + n0;
            int r   = rowBase + mofs + j;
            if (col < N && r < M) Cout[(size_t)r * ldC + col] = acc[nt][j];
        }
}

// ---- small utility kernels --------------------------------------------------
__global__ void k_fill(float* p, int n, float v) {
    int t = blockIdx.x * blockDim.x + threadIdx.x;
    if (t < n) p[t] = v;
}
__global__ void k_count(const int* __restrict__ dst, int E, float* deg) {
    int t = blockIdx.x * blockDim.x + threadIdx.x;
    if (t < E) atomicAdd(&deg[dst[t]], 1.f);
}
__global__ void k_rsqrt1p(float* p, int n) {
    int t = blockIdx.x * blockDim.x + threadIdx.x;
    if (t < n) p[t] = rsqrtf(1.f + p[t]);
}
// out[i,c] = bias[c] + h[i,c]*dinv[i]^2
__global__ void k_init(const float* __restrict__ h, const float* __restrict__ dinv,
                       const float* __restrict__ bias, int M, int C, float* __restrict__ outp) {
    int t = blockIdx.x * blockDim.x + threadIdx.x;
    if (t < M * C) {
        int i = t / C, c = t - i * C;
        float dv = dinv[i];
        outp[t] = bias[c] + h[t] * dv * dv;
    }
}
// scatter: out[dst] += h[src]*dinv[src]*dinv[dst]  (4-channel chunks)
__global__ void k_edge(const int* __restrict__ src, const int* __restrict__ dst, int E,
                       const float* __restrict__ h, const float* __restrict__ dinv,
                       int C, int CC, float* __restrict__ outp) {
    int t = blockIdx.x * blockDim.x + threadIdx.x;
    if (t >= E * CC) return;
    int e = t / CC, cc = t - e * CC;
    int s = src[e], d = dst[e];
    float w = dinv[s] * dinv[d];
    int c0 = cc * 4;
    for (int c = c0; c < c0 + 4 && c < C; ++c)
        atomicAdd(&outp[(size_t)d * C + c], h[(size_t)s * C + c] * w);
}

// ---- theta = kerW kerW^T ; A = softmax(causal-masked relu(M M^T)) -----------
__global__ __launch_bounds__(256) void k_prep(const float* __restrict__ kerW,
                                              const float* __restrict__ Mm,
                                              float* __restrict__ theta,
                                              float* __restrict__ Aout) {
    __shared__ float sM[80 * 26];
    __shared__ float Ar[80 * 80];
    int tid = threadIdx.x;
    for (int e = tid; e < 80 * 26; e += 256) sM[e] = Mm[e];
    __syncthreads();
    for (int e = tid; e < 6400; e += 256) {
        int i = e / 80, j = e - i * 80;
        if (j > i) { Ar[e] = -__builtin_inff(); continue; }
        float s = 0.f;
        for (int k = 0; k < 26; ++k) s += sM[i * 26 + k] * sM[j * 26 + k];
        Ar[e] = fmaxf(s, 0.f);
    }
    __syncthreads();
    if (tid < 80) {
        float mx = -__builtin_inff();
        for (int j = 0; j < 80; ++j) mx = fmaxf(mx, Ar[tid * 80 + j]);
        float sum = 0.f;
        for (int j = 0; j < 80; ++j) sum += expf(Ar[tid * 80 + j] - mx);
        float inv = 1.f / sum;
        for (int j = 0; j < 80; ++j)
            Aout[tid * 80 + j] = expf(Ar[tid * 80 + j] - mx) * inv;
    }
    for (int e = tid; e < 34 * 34; e += 256) {
        int i = e / 34, j = e - i * 34;
        float s = 0.f;
        for (int k = 0; k < 128; ++k) s += kerW[i * 128 + k] * kerW[j * 128 + k];
        theta[e] = s;  // already symmetric
    }
}

// ---- kernel attention: alpha = softmax(-0.5 dThetad / s); yh = alpha @ y ----
__global__ __launch_bounds__(256) void k_attn(
    const float* __restrict__ x1k, const float* __restrict__ xpre1,
    const float* __restrict__ xpre2, const float* __restrict__ theta,
    const float* __restrict__ Aout, const float* __restrict__ smoothing,
    float* __restrict__ x2mod)
{
    int b = blockIdx.x, tid = threadIdx.x;
    __shared__ float xp[64][34], xf[16][34], th[34][34], Gp[64][34];
    __shared__ float sp[64], sf[16], yk[64], al[16][64];

    for (int e = tid; e < 64 * 34; e += 256) { int p = e / 34, c = e - p * 34; xp[p][c] = x1k[((size_t)b * 64 + p) * 34 + c]; }
    for (int e = tid; e < 16 * 34; e += 256) { int q = e / 34, c = e - q * 34; xf[q][c] = xpre2[((size_t)b * 16 + q) * 35 + c]; }
    for (int e = tid; e < 34 * 34; e += 256) th[e / 34][e % 34] = theta[e];
    if (tid < 64) yk[tid] = xpre1[((size_t)b * 64 + tid) * 35 + 34];
    __syncthreads();

    for (int e = tid; e < 64 * 34; e += 256) {
        int p = e / 34, o = e - p * 34;
        float s = 0.f;
        for (int k = 0; k < 34; ++k) s += xp[p][k] * th[k][o];
        Gp[p][o] = s;
    }
    __syncthreads();
    if (tid < 64) {
        float s = 0.f;
        for (int c = 0; c < 34; ++c) s += xp[tid][c] * Gp[tid][c];
        sp[tid] = s;
    }
    if (tid < 16) {
        float s = 0.f;
        for (int o = 0; o < 34; ++o) {
            float g = 0.f;
            for (int k = 0; k < 34; ++k) g += th[o][k] * xf[tid][k];
            s += xf[tid][o] * g;
        }
        sf[tid] = s;
    }
    __syncthreads();

    float sg = 1.f / (1.f + expf(-smoothing[0]));
    float scale = -0.5f / (sg * 0.01f);
    for (int e = tid; e < 16 * 64; e += 256) {
        int q = e >> 6, p = e & 63;
        float cr = 0.f;
        for (int k = 0; k < 34; ++k) cr += xf[q][k] * Gp[p][k];
        float a = scale * (sp[p] + sf[q] - 2.f * cr);
        if (Aout[(64 + q) * 80 + p] == 0.f) a = -__builtin_inff();
        al[q][p] = a;
    }
    __syncthreads();

    if (tid < 16) {
        float mx = -__builtin_inff();
        for (int p = 0; p < 64; ++p) mx = fmaxf(mx, al[tid][p]);
        float sum = 0.f;
        for (int p = 0; p < 64; ++p) { float e = expf(al[tid][p] - mx); al[tid][p] = e; sum += e; }
        float inv = 1.f / sum;
        float yh = 0.f;
        for (int p = 0; p < 64; ++p) yh += al[tid][p] * inv * yk[p];
        x2mod[((size_t)b * 16 + tid) * 35 + 34] = yh;
    }
    for (int e = tid; e < 16 * 34; e += 256) {
        int q = e / 34, c = e - q * 34;
        x2mod[((size_t)b * 16 + q) * 35 + c] = xf[q][c];
    }
}

// ---- per-batch 80x80 pairwise distances ------------------------------------
__global__ __launch_bounds__(256) void k_cd(const float* __restrict__ xpre1,
                                            const float* __restrict__ x2mod,
                                            float* __restrict__ cdout) {
    int b = blockIdx.x, tid = threadIdx.x;
    __shared__ float xa[80][35];
    __shared__ float sq[80];
    for (int e = tid; e < 80 * 35; e += 256) {
        int r = e / 35, c = e - r * 35;
        xa[r][c] = (r < 64) ? xpre1[((size_t)b * 64 + r) * 35 + c]
                            : x2mod[((size_t)b * 16 + (r - 64)) * 35 + c];
    }
    __syncthreads();
    if (tid < 80) {
        float s = 0.f;
        for (int c = 0; c < 35; ++c) s += xa[tid][c] * xa[tid][c];
        sq[tid] = s;
    }
    __syncthreads();
    for (int e = tid; e < 6400; e += 256) {
        int i = e / 80, j = e - i * 80;
        float d = 0.f;
        for (int c = 0; c < 35; ++c) d += xa[i][c] * xa[j][c];
        float d2 = sq[i] + sq[j] - 2.f * d;
        cdout[(size_t)b * 6400 + e] = sqrtf(fmaxf(d2, 0.f));
    }
}

static inline int cdiv(long long a, int b) { return (int)((a + b - 1) / b); }

extern "C" void kernel_launch(void* const* d_in, const int* in_sizes, int n_in,
                              void* d_out, int out_size, void* d_ws, size_t ws_size,
                              hipStream_t stream)
{
    const int*   x1_cat = (const int*)d_in[0];
    const float* x1_num = (const float*)d_in[1];
    const int*   x2_cat = (const int*)d_in[2];
    const float* x2_num = (const float*)d_in[3];
    const int*   e1     = (const int*)d_in[4];
    const int*   e2     = (const int*)d_in[5];
    const float* emb0   = (const float*)d_in[6];
    const float* emb1   = (const float*)d_in[7];
    const float* emb2   = (const float*)d_in[8];
    const float* pW1 = (const float*)d_in[9],  *pb1 = (const float*)d_in[10];
    const float* pW2 = (const float*)d_in[11], *pb2 = (const float*)d_in[12];
    const float* pW3 = (const float*)d_in[13], *pb3 = (const float*)d_in[14];
    const float* g1W1 = (const float*)d_in[15], *g1b1 = (const float*)d_in[16];
    const float* g1W2 = (const float*)d_in[17], *g1b2 = (const float*)d_in[18];
    const float* g2W1 = (const float*)d_in[19], *g2b1 = (const float*)d_in[20];
    const float* g2W2 = (const float*)d_in[21], *g2b2 = (const float*)d_in[22];
    const float* kerW = (const float*)d_in[23];
    const float* smoothing = (const float*)d_in[24];
    const float* Mm   = (const float*)d_in[25];

    const int n1 = SBATCH * PASTN;   // 131072
    const int n2 = SBATCH * FUTN;    // 32768
    const int E1 = in_sizes[4] / 2;
    const int E2 = in_sizes[5] / 2;

    float* ws    = (float*)d_ws;
    float* xpre1 = ws;
    float* xpre2 = xpre1 + (size_t)n1 * 35;
    float* theta = xpre2 + (size_t)n2 * 35;
    float* dinv1 = theta + 34 * 34;
    float* dinv2 = dinv1 + n1;
    float* h1    = dinv2 + n2;
    float* agg1  = h1 + (size_t)n1 * 64;
    float* h1k   = agg1 + (size_t)n1 * 64;
    float* x1k   = h1k + (size_t)n1 * 34;
    float* x2mod = x1k + (size_t)n1 * 34;
    float* h2 = h1, *agg2 = agg1, *hsc = h1k;   // reuse (graph1 buffers dead by then)

    float* out0  = (float*)d_out;
    float* cdout = out0 + (size_t)SBATCH * FUTN;
    float* Aout  = cdout + (size_t)SBATCH * 80 * 80;

    // theta + A
    k_prep<<<1, 256, 0, stream>>>(kerW, Mm, theta, Aout);

    // degree -> dinv (per graph)
    k_fill<<<cdiv(n1, 256), 256, 0, stream>>>(dinv1, n1, 0.f);
    k_fill<<<cdiv(n2, 256), 256, 0, stream>>>(dinv2, n2, 0.f);
    k_count<<<cdiv(E1, 256), 256, 0, stream>>>(e1 + E1, E1, dinv1);
    k_count<<<cdiv(E2, 256), 256, 0, stream>>>(e2 + E2, E2, dinv2);
    k_rsqrt1p<<<cdiv(n1, 256), 256, 0, stream>>>(dinv1, n1);
    k_rsqrt1p<<<cdiv(n2, 256), 256, 0, stream>>>(dinv2, n2);

    // fused pre-MLP (WMMA)
    k_mlp<<<n1 / 128, 256, 0, stream>>>(x1_cat, x1_num, emb0, emb1, emb2,
                                        pW1, pb1, pW2, pb2, pW3, pb3, xpre1, n1);
    k_mlp<<<n2 / 128, 256, 0, stream>>>(x2_cat, x2_num, emb0, emb1, emb2,
                                        pW1, pb1, pW2, pb2, pW3, pb3, xpre2, n2);

    // GCN1 layer1 (35->64), relu folded into next GEMM's read
    k_gemm<2, 4, 0><<<n1 / 128, 256, 0, stream>>>(xpre1, n1, 35, 35, g1W1, 64, h1, 64);
    k_init<<<cdiv((long long)n1 * 64, 256), 256, 0, stream>>>(h1, dinv1, g1b1, n1, 64, agg1);
    k_edge<<<cdiv((long long)E1 * 16, 256), 256, 0, stream>>>(e1, e1 + E1, E1, h1, dinv1, 64, 16, agg1);
    // GCN1 layer2 (64->34)
    k_gemm<2, 3, 1><<<n1 / 128, 256, 0, stream>>>(agg1, n1, 64, 64, g1W2, 34, h1k, 34);
    k_init<<<cdiv((long long)n1 * 34, 256), 256, 0, stream>>>(h1k, dinv1, g1b2, n1, 34, x1k);
    k_edge<<<cdiv((long long)E1 * 9, 256), 256, 0, stream>>>(e1, e1 + E1, E1, h1k, dinv1, 34, 9, x1k);

    // kernel attention -> x2mod (34 ch copy + yh in ch 34)
    k_attn<<<SBATCH, 256, 0, stream>>>(x1k, xpre1, xpre2, theta, Aout, smoothing, x2mod);

    // GCN2 layer1 (35->64)
    k_gemm<2, 4, 0><<<n2 / 128, 256, 0, stream>>>(x2mod, n2, 35, 35, g2W1, 64, h2, 64);
    k_init<<<cdiv((long long)n2 * 64, 256), 256, 0, stream>>>(h2, dinv2, g2b1, n2, 64, agg2);
    k_edge<<<cdiv((long long)E2 * 16, 256), 256, 0, stream>>>(e2, e2 + E2, E2, h2, dinv2, 64, 16, agg2);
    // GCN2 layer2 (64->1) -> final `out` section of d_out
    k_gemm<2, 1, 1><<<n2 / 128, 256, 0, stream>>>(agg2, n2, 64, 64, g2W2, 1, hsc, 1);
    k_init<<<cdiv(n2, 256), 256, 0, stream>>>(hsc, dinv2, g2b2, n2, 1, out0);
    k_edge<<<cdiv(E2, 256), 256, 0, stream>>>(e2, e2 + E2, E2, hsc, dinv2, 1, 1, out0);

    // pairwise distances
    k_cd<<<SBATCH, 256, 0, stream>>>(xpre1, x2mod, cdout);

    (void)n_in; (void)out_size; (void)ws_size;
}
